// GAE_27728308863310
// MI455X (gfx1250) — matrix operand order repaired and checked by reading, not compiled, not run on hardware
//
#include <hip/hip_runtime.h>
#include <hip/hip_bf16.h>

#define NNODES 50000
#define NEDGES 1200000
#define NGRAPH 256

typedef __attribute__((ext_vector_type(16))) _Float16 v16h;
typedef __attribute__((ext_vector_type(8)))  float    v8f;

__device__ __forceinline__ float siluf(float x){ return x * (1.f/(1.f+__expf(-x))); }
__device__ __forceinline__ float sigmf(float x){ return 1.f/(1.f+__expf(-x)); }
__device__ __forceinline__ void lds_wave_fence(){ asm volatile("s_wait_dscnt 0" ::: "memory"); }

// ---------------- WMMA fragment helpers (wave32, f16 in / f32 acc) ----------------
// A-matrix 16x32 f16 per ISA layout: lanes 0-15 M=lane (K groups 0..7,16..23),
// lanes 16-31 M=lane-16 (K groups 8..15,24..31).
__device__ __forceinline__ v16h frag_a(const _Float16* A, int kpad, int k0, int lane){
  int m = lane & 15, half = lane >> 4;
  const _Float16* row = A + m*kpad + k0;
  v16h f;
  #pragma unroll
  for (int v2 = 0; v2 < 8; ++v2){
    int k = (v2 < 4) ? (half*8 + 2*v2) : (16 + half*8 + 2*(v2-4));
    f[2*v2]   = row[k];
    f[2*v2+1] = row[k+1];
  }
  return f;
}
// B-matrix 32x16 f16: lane n=L%16, half=L/16 holds K groups half*8..+7 and 16+half*8..+7 of column n (same as A).
// Bt is stored column-major-in-K in LDS: Bt[col*kpad + k] -> contiguous 32b reads.
__device__ __forceinline__ v16h frag_b(const _Float16* Bt, int kpad, int colBase, int k0, int lane){
  int n = lane & 15, half = lane >> 4;
  const _Float16* cp = Bt + (colBase + n)*kpad + k0;
  v16h f;
  #pragma unroll
  for (int v2 = 0; v2 < 8; ++v2){ int k = (v2 < 4) ? (half*8 + 2*v2) : (16 + half*8 + 2*(v2-4)); f[2*v2] = cp[k]; f[2*v2+1] = cp[k+1]; }
  return f;
}

template<int NT>
__device__ __forceinline__ void zacc(v8f* a){
  #pragma unroll
  for (int i = 0; i < NT; ++i){
    #pragma unroll
    for (int j = 0; j < 8; ++j) a[i][j] = 0.f;
  }
}

template<int KPAD, int NT>
__device__ __forceinline__ void wave_gemm(const _Float16* A, const _Float16* Bt, v8f* acc, int lane){
  #pragma unroll
  for (int kc = 0; kc < KPAD/32; ++kc){
    v16h a = frag_a(A, KPAD, kc*32, lane);
    #pragma unroll
    for (int nt = 0; nt < NT; ++nt){
      v16h b = frag_b(Bt, KPAD, nt*16, kc*32, lane);
      acc[nt] = __builtin_amdgcn_wmma_f32_16x16x32_f16(
          false, a, false, b, (short)0, acc[nt], false, false);
    }
  }
}

// Stage W[K][Mout] (f32, row-major) -> LDS Bt[MoutPad][kpad] f16, zero-padded.
__device__ __forceinline__ void stage_w2(const float* W, _Float16* Bt, int K, int Mout,
                                         int MoutPad, int kpad, int tid, int nth){
  for (int i = tid; i < MoutPad*kpad; i += nth){
    int col = i / kpad, k = i - col*kpad;
    Bt[i] = (k < K && col < Mout) ? (_Float16)W[k*Mout + col] : (_Float16)0.f;
  }
}

// C/D layout: VGPR r -> row r + 8*(lane/16), col = lane%16 (+16*nt).
__device__ __forceinline__ void acc_to_ldsA(const v8f* acc, int ntc, const float* bias,
                                            _Float16* T, int tpad, int lane){
  int n = lane & 15, half = lane >> 4;
  for (int nt = 0; nt < ntc; ++nt){
    int col = nt*16 + n;
    float b = bias[col];
    #pragma unroll
    for (int r = 0; r < 8; ++r)
      T[(r + 8*half)*tpad + col] = (_Float16)siluf(acc[nt][r] + b);
  }
}

__device__ __forceinline__ void acc_store(const v8f* acc, int ntc, const float* bias,
                                          float* C, long ldc, long m0, long rmax,
                                          int lane, int do_silu){
  int n = lane & 15, half = lane >> 4;
  for (int nt = 0; nt < ntc; ++nt){
    int col = nt*16 + n;
    float b = bias[col];
    #pragma unroll
    for (int r = 0; r < 8; ++r){
      long row = m0 + r + 8*half;
      if (row < rmax){
        float v = acc[nt][r] + b;
        C[row*ldc + col] = do_silu ? siluf(v) : v;
      }
    }
  }
}

// ---------------- Stage 1: node encoder h = silu([x|feats] @ W_node + b) ----------------
__global__ void __launch_bounds__(128)
k_node_enc(const float* __restrict__ x, const float* __restrict__ feats,
           const float* __restrict__ Wn, const float* __restrict__ bn,
           float* __restrict__ hout){
  __shared__ _Float16 sB[64*160];
  __shared__ _Float16 sA[4][16*160];
  int tid = threadIdx.x, lane = tid & 31, w = tid >> 5;
  stage_w2(Wn, sB, 130, 64, 64, 160, tid, 128);
  long m0 = ((long)blockIdx.x*4 + w)*16;
  _Float16* A = sA[w];
  for (int i = lane; i < 16*160; i += 32){
    int r = i/160, c = i - r*160;
    long row = m0 + r;
    float v = 0.f;
    if (row < NNODES){
      if (c < 119) v = x[row*119 + c];
      else if (c < 130) v = feats[row*11 + (c-119)];
    }
    A[i] = (_Float16)v;
  }
  __syncthreads();
  v8f acc[4]; zacc<4>(acc);
  wave_gemm<160,4>(A, sB, acc, lane);
  acc_store(acc, 4, bn, hout, 64, m0, NNODES, lane, 1);
}

// ---------------- Stage 2: edge encoder e = silu(ea@We1+b1)@We2+b2 (fused) ----------------
__global__ void __launch_bounds__(128)
k_edge_enc(const float* __restrict__ ea, const float* __restrict__ We1, const float* __restrict__ be1,
           const float* __restrict__ We2, const float* __restrict__ be2, float* __restrict__ eout){
  __shared__ _Float16 sB1[64*32];
  __shared__ _Float16 sB2[64*64];
  __shared__ _Float16 sA[4][16*32];
  __shared__ _Float16 sT[4][16*64];
  int tid = threadIdx.x, lane = tid & 31, w = tid >> 5;
  stage_w2(We1, sB1, 4, 64, 64, 32, tid, 128);
  stage_w2(We2, sB2, 64, 64, 64, 64, tid, 128);
  long m0 = ((long)blockIdx.x*4 + w)*16;
  _Float16* A = sA[w];
  for (int i = lane; i < 16*32; i += 32){
    int r = i >> 5, c = i & 31;
    long row = m0 + r;
    A[i] = (_Float16)((row < NEDGES && c < 4) ? ea[row*4 + c] : 0.f);
  }
  __syncthreads();
  v8f acc[4]; zacc<4>(acc);
  wave_gemm<32,4>(A, sB1, acc, lane);
  acc_to_ldsA(acc, 4, be1, sT[w], 64, lane);
  lds_wave_fence();
  zacc<4>(acc);
  wave_gemm<64,4>(sT[w], sB2, acc, lane);
  acc_store(acc, 4, be2, eout, 64, m0, NEDGES, lane, 0);
}

// ---------------- GINE message: agg[dst] += relu(h[src] + e) ----------------
__global__ void __launch_bounds__(256)
k_message(const float* __restrict__ h, const float* __restrict__ e,
          const int* __restrict__ ei, float* __restrict__ agg){
  long idx = (long)blockIdx.x*256 + threadIdx.x;           // E*16 threads, 4 floats each
  if (idx >= (long)NEDGES*16) return;
  int edge = (int)(idx >> 4), part = (int)(idx & 15);
  int src = ei[edge], dst = ei[NEDGES + edge];
  float4 ev = ((const float4*)(e + (long)edge*64))[part];
  float4 hv = ((const float4*)(h + (long)src*64))[part];
  float4 m;
  m.x = fmaxf(ev.x + hv.x, 0.f); m.y = fmaxf(ev.y + hv.y, 0.f);
  m.z = fmaxf(ev.z + hv.z, 0.f); m.w = fmaxf(ev.w + hv.w, 0.f);
  float* ap = agg + (long)dst*64 + part*4;
  atomicAdd(ap+0, m.x); atomicAdd(ap+1, m.y); atomicAdd(ap+2, m.z); atomicAdd(ap+3, m.w);
}

// ---------------- GINE MLP: z2 = silu((h+agg)@W1+b1)@W2+b2, plus column sum/sumsq ----------------
__global__ void __launch_bounds__(128)
k_gine_mlp(const float* __restrict__ h, const float* __restrict__ agg,
           const float* __restrict__ W1, const float* __restrict__ b1,
           const float* __restrict__ W2, const float* __restrict__ b2,
           float* __restrict__ z2, float* __restrict__ colsum, float* __restrict__ colsq){
  __shared__ _Float16 sB1[64*64];
  __shared__ _Float16 sB2[64*64];
  __shared__ _Float16 sA[4][16*64];
  __shared__ _Float16 sT[4][16*64];
  int tid = threadIdx.x, lane = tid & 31, w = tid >> 5;
  stage_w2(W1, sB1, 64, 64, 64, 64, tid, 128);
  stage_w2(W2, sB2, 64, 64, 64, 64, tid, 128);
  long m0 = ((long)blockIdx.x*4 + w)*16;
  _Float16* A = sA[w];
  for (int i = lane; i < 16*64; i += 32){
    int r = i >> 6, c = i & 63;
    long row = m0 + r;
    A[i] = (_Float16)((row < NNODES) ? (h[row*64 + c] + agg[row*64 + c]) : 0.f);
  }
  __syncthreads();
  v8f acc[4]; zacc<4>(acc);
  wave_gemm<64,4>(A, sB1, acc, lane);
  acc_to_ldsA(acc, 4, b1, sT[w], 64, lane);
  lds_wave_fence();
  zacc<4>(acc);
  wave_gemm<64,4>(sT[w], sB2, acc, lane);
  int n = lane & 15, half = lane >> 4;
  #pragma unroll
  for (int nt = 0; nt < 4; ++nt){
    int col = nt*16 + n;
    float b = b2[col], s = 0.f, s2 = 0.f;
    #pragma unroll
    for (int r = 0; r < 8; ++r){
      long row = m0 + r + 8*half;
      if (row < NNODES){
        float v = acc[nt][r] + b;
        z2[row*64 + col] = v;
        s += v; s2 += v*v;
      }
    }
    atomicAdd(&colsum[col], s);
    atomicAdd(&colsq[col], s2);
  }
}

__global__ void k_bn_fin(const float* __restrict__ colsum, const float* __restrict__ colsq,
                         const float* __restrict__ g, const float* __restrict__ b,
                         float* __restrict__ scale, float* __restrict__ shift){
  int c = threadIdx.x;
  float mu  = colsum[c] / (float)NNODES;
  float var = colsq[c] / (float)NNODES - mu*mu;
  float inv = rsqrtf(var + 1e-5f);
  float sc  = g[c]*inv;
  scale[c] = sc;
  shift[c] = b[c] - mu*sc;
}

__global__ void __launch_bounds__(256)
k_bn_apply(const float* __restrict__ z2, const float* __restrict__ scale,
           const float* __restrict__ shift, float* __restrict__ hout){
  long i = (long)blockIdx.x*256 + threadIdx.x;
  if (i < (long)NNODES*64){
    int c = (int)(i & 63);
    hout[i] = siluf(z2[i]*scale[c] + shift[c]);
  }
}

// ---------------- Set2Set ----------------
__global__ void __launch_bounds__(256)
k_lstm_gates(const float* __restrict__ qstar, const float* __restrict__ hs,
             const float* __restrict__ Wih, const float* __restrict__ Whh,
             const float* __restrict__ bl, float* __restrict__ gates){
  int b = blockIdx.x, c = threadIdx.x;
  float g = bl[c];
  const float* q = qstar + b*128;
  #pragma unroll 4
  for (int k = 0; k < 128; ++k) g += q[k]*Wih[k*256 + c];
  const float* hh = hs + b*64;
  #pragma unroll 4
  for (int k = 0; k < 64; ++k)  g += hh[k]*Whh[k*256 + c];
  gates[b*256 + c] = g;
}

__global__ void __launch_bounds__(256)
k_lstm_update(const float* __restrict__ gates, float* __restrict__ cs, float* __restrict__ hs){
  int i = blockIdx.x*256 + threadIdx.x;
  if (i >= NGRAPH*64) return;
  int b = i >> 6, c = i & 63;
  const float* g = gates + b*256;
  float gi = sigmf(g[c]), gf = sigmf(g[64+c]), gg = tanhf(g[128+c]), go = sigmf(g[192+c]);
  float cn = gf*cs[i] + gi*gg;
  cs[i] = cn;
  hs[i] = go*tanhf(cn);
}

__global__ void __launch_bounds__(256)
k_att(const float* __restrict__ h, const float* __restrict__ hs, const int* __restrict__ batch,
      float* __restrict__ att, unsigned* __restrict__ mkey){
  int n = blockIdx.x*256 + threadIdx.x;
  if (n >= NNODES) return;
  int b = batch[n];
  const float4* hp = (const float4*)(h + (long)n*64);
  const float4* q4 = (const float4*)(hs + (long)b*64);
  float s = 0.f;
  #pragma unroll
  for (int i = 0; i < 16; ++i){
    float4 a = hp[i], bq = q4[i];
    s += a.x*bq.x + a.y*bq.y + a.z*bq.z + a.w*bq.w;
  }
  att[n] = s;
  unsigned u = __float_as_uint(s);
  u = (u & 0x80000000u) ? ~u : (u | 0x80000000u);   // order-preserving key
  atomicMax(&mkey[b], u);
}

__global__ void __launch_bounds__(256)
k_wden(const float* __restrict__ att, const int* __restrict__ batch,
       const unsigned* __restrict__ mkey, float* __restrict__ wexp, float* __restrict__ den){
  int n = blockIdx.x*256 + threadIdx.x;
  if (n >= NNODES) return;
  int b = batch[n];
  unsigned u = mkey[b];
  unsigned b0 = (u & 0x80000000u) ? (u ^ 0x80000000u) : ~u;
  float m = __uint_as_float(b0);
  float v = __expf(att[n] - m);
  wexp[n] = v;
  atomicAdd(&den[b], v);
}

__global__ void __launch_bounds__(256)
k_rsum(const float* __restrict__ h, const float* __restrict__ wexp, const float* __restrict__ den,
       const int* __restrict__ batch, float* __restrict__ rbuf){
  long idx = (long)blockIdx.x*256 + threadIdx.x;
  if (idx >= (long)NNODES*16) return;
  int n = (int)(idx >> 4), part = (int)(idx & 15);
  int b = batch[n];
  float a = wexp[n] / den[b];
  const float* hp = h + (long)n*64 + part*4;
  float* rp = rbuf + (long)b*64 + part*4;
  atomicAdd(rp+0, a*hp[0]); atomicAdd(rp+1, a*hp[1]);
  atomicAdd(rp+2, a*hp[2]); atomicAdd(rp+3, a*hp[3]);
}

__global__ void __launch_bounds__(256)
k_qstar(const float* __restrict__ hs, const float* __restrict__ rbuf, float* __restrict__ qstar){
  int i = blockIdx.x*256 + threadIdx.x;
  if (i >= NGRAPH*128) return;
  int b = i >> 7, c = i & 127;
  qstar[i] = (c < 64) ? hs[b*64 + c] : rbuf[b*64 + (c - 64)];
}

// ---------------- latent = silu(qstar@Wg1+bg1)@Wg2+bg2 ----------------
__global__ void __launch_bounds__(128)
k_graph_mlp(const float* __restrict__ qstar, const float* __restrict__ Wg1, const float* __restrict__ bg1,
            const float* __restrict__ Wg2, const float* __restrict__ bg2, float* __restrict__ latent){
  __shared__ _Float16 sB1[64*128];
  __shared__ _Float16 sB2[64*64];
  __shared__ _Float16 sA[4][16*128];
  __shared__ _Float16 sT[4][16*64];
  int tid = threadIdx.x, lane = tid & 31, w = tid >> 5;
  stage_w2(Wg1, sB1, 128, 64, 64, 128, tid, 128);
  stage_w2(Wg2, sB2, 64, 64, 64, 64, tid, 128);
  long m0 = ((long)blockIdx.x*4 + w)*16;
  _Float16* A = sA[w];
  for (int i = lane; i < 16*128; i += 32){
    int r = i >> 7, c = i & 127;
    long row = m0 + r;
    A[i] = (_Float16)((row < NGRAPH) ? qstar[row*128 + c] : 0.f);
  }
  __syncthreads();
  v8f acc[4]; zacc<4>(acc);
  wave_gemm<128,4>(A, sB1, acc, lane);
  acc_to_ldsA(acc, 4, bg1, sT[w], 64, lane);
  lds_wave_fence();
  zacc<4>(acc);
  wave_gemm<64,4>(sT[w], sB2, acc, lane);
  acc_store(acc, 4, bg2, latent, 64, m0, NGRAPH, lane, 0);
}

// ---------------- node_recon = silu(gn@Wn1+bn1)@Wn2+bn2, gn = latent[batch] ----------------
__global__ void __launch_bounds__(128)
k_node_recon(const float* __restrict__ latent, const int* __restrict__ batch,
             const float* __restrict__ Wn1, const float* __restrict__ bn1,
             const float* __restrict__ Wn2, const float* __restrict__ bn2,
             float* __restrict__ outp){
  __shared__ _Float16 sB1[32*64];     // Bt[32][64]
  __shared__ _Float16 sB2[128*32];    // Bt[128][32], cols>=119 zero
  __shared__ _Float16 sA[4][16*64];
  __shared__ _Float16 sT[4][16*32];
  int tid = threadIdx.x, lane = tid & 31, w = tid >> 5;
  stage_w2(Wn1, sB1, 64, 32, 32, 64, tid, 128);
  stage_w2(Wn2, sB2, 32, 119, 128, 32, tid, 128);
  long m0 = ((long)blockIdx.x*4 + w)*16;
  _Float16* A = sA[w];
  for (int i = lane; i < 16*64; i += 32){
    int r = i >> 6, c = i & 63;
    long row = m0 + r;
    float v = 0.f;
    if (row < NNODES){ int b = batch[row]; v = latent[(long)b*64 + c]; }
    A[i] = (_Float16)v;
  }
  __syncthreads();
  v8f acc1[2]; zacc<2>(acc1);
  wave_gemm<64,2>(A, sB1, acc1, lane);
  acc_to_ldsA(acc1, 2, bn1, sT[w], 32, lane);
  lds_wave_fence();
  v8f acc2[8]; zacc<8>(acc2);
  wave_gemm<32,8>(sT[w], sB2, acc2, lane);
  int n = lane & 15, half = lane >> 4;
  for (int nt = 0; nt < 8; ++nt){
    int col = nt*16 + n;
    if (col < 119){
      float b = bn2[col];
      #pragma unroll
      for (int r = 0; r < 8; ++r){
        long row = m0 + r + 8*half;
        if (row < NNODES) outp[row*119 + col] = acc2[nt][r] + b;
      }
    }
  }
}

// ---------------- edge_in staging helper ----------------
__device__ __forceinline__ void stage_edge_in(_Float16* A, const float* latent, const int* batch,
                                              const int* ei, long m0, int lane){
  for (int i = lane; i < 16*128; i += 32){
    int r = i >> 7, c = i & 127;
    long row = m0 + r;
    float v = 0.f;
    if (row < NEDGES){
      int node = (c < 64) ? ei[row] : ei[NEDGES + row];
      int b = batch[node];
      v = latent[(long)b*64 + (c & 63)];
    }
    A[i] = (_Float16)v;
  }
}

// ---------------- edge_logits = silu(edge_in@Wp1+bp1)@Wp2+bp2 ----------------
__global__ void __launch_bounds__(128)
k_edge_logit(const float* __restrict__ latent, const int* __restrict__ batch, const int* __restrict__ ei,
             const float* __restrict__ Wp1, const float* __restrict__ bp1,
             const float* __restrict__ Wp2, const float* __restrict__ bp2,
             float* __restrict__ outp){
  __shared__ _Float16 sB1[64*128];
  __shared__ _Float16 sB2[16*64];
  __shared__ _Float16 sA[4][16*128];
  __shared__ _Float16 sT[4][16*64];
  int tid = threadIdx.x, lane = tid & 31, w = tid >> 5;
  stage_w2(Wp1, sB1, 128, 64, 64, 128, tid, 128);
  stage_w2(Wp2, sB2, 64, 1, 16, 64, tid, 128);
  long m0 = ((long)blockIdx.x*4 + w)*16;
  stage_edge_in(sA[w], latent, batch, ei, m0, lane);
  __syncthreads();
  v8f acc[4]; zacc<4>(acc);
  wave_gemm<128,4>(sA[w], sB1, acc, lane);
  acc_to_ldsA(acc, 4, bp1, sT[w], 64, lane);
  lds_wave_fence();
  v8f acc2[1]; zacc<1>(acc2);
  wave_gemm<64,1>(sT[w], sB2, acc2, lane);
  int n = lane & 15, half = lane >> 4;
  if (n == 0){
    float b = bp2[0];
    #pragma unroll
    for (int r = 0; r < 8; ++r){
      long row = m0 + r + 8*half;
      if (row < NEDGES) outp[row] = acc2[0][r] + b;
    }
  }
}

// ---------------- edge_recon = sp/lin(silu(silu(edge_in@Wd1+bd1)@Wd2+bd2)@Wd3+bd3) ----------------
__global__ void __launch_bounds__(128)
k_edge_recon(const float* __restrict__ latent, const int* __restrict__ batch, const int* __restrict__ ei,
             const float* __restrict__ Wd1, const float* __restrict__ bd1,
             const float* __restrict__ Wd2, const float* __restrict__ bd2,
             const float* __restrict__ Wd3, const float* __restrict__ bd3,
             float* __restrict__ outp){
  __shared__ _Float16 sB1[64*128];
  __shared__ _Float16 sB2[64*64];
  __shared__ _Float16 sB3[16*64];
  __shared__ _Float16 sA[4][16*128];
  __shared__ _Float16 sT[4][16*64];
  int tid = threadIdx.x, lane = tid & 31, w = tid >> 5;
  stage_w2(Wd1, sB1, 128, 64, 64, 128, tid, 128);
  stage_w2(Wd2, sB2, 64, 64, 64, 64, tid, 128);
  stage_w2(Wd3, sB3, 64, 4, 16, 64, tid, 128);
  long m0 = ((long)blockIdx.x*4 + w)*16;
  stage_edge_in(sA[w], latent, batch, ei, m0, lane);
  __syncthreads();
  v8f acc[4]; zacc<4>(acc);
  wave_gemm<128,4>(sA[w], sB1, acc, lane);
  acc_to_ldsA(acc, 4, bd1, sT[w], 64, lane);
  lds_wave_fence();
  zacc<4>(acc);
  wave_gemm<64,4>(sT[w], sB2, acc, lane);
  lds_wave_fence();                       // t1 reads done before overwrite
  acc_to_ldsA(acc, 4, bd2, sT[w], 64, lane);
  lds_wave_fence();
  v8f acc3[1]; zacc<1>(acc3);
  wave_gemm<64,1>(sT[w], sB3, acc3, lane);
  int n = lane & 15, half = lane >> 4;
  if (n < 4){
    float b = bd3[n];
    #pragma unroll
    for (int r = 0; r < 8; ++r){
      long row = m0 + r + 8*half;
      if (row < NEDGES){
        float v = acc3[0][r] + b;
        if (n == 0) v = (v > 20.f) ? v : log1pf(__expf(v));   // softplus on first channel
        outp[row*4 + n] = v;
      }
    }
  }
}

__global__ void __launch_bounds__(256)
k_zero(float* __restrict__ p, long n){
  long i = (long)blockIdx.x*256 + threadIdx.x;
  if (i < n) p[i] = 0.f;
}

// ---------------- host launch ----------------
extern "C" void kernel_launch(void* const* d_in, const int* in_sizes, int n_in,
                              void* d_out, int out_size, void* d_ws, size_t ws_size,
                              hipStream_t stream) {
  const float* x     = (const float*)d_in[0];
  const float* feats = (const float*)d_in[1];
  const float* ea    = (const float*)d_in[2];
  const float* Wnode = (const float*)d_in[3];  const float* bnode = (const float*)d_in[4];
  const float* We1   = (const float*)d_in[5];  const float* be1   = (const float*)d_in[6];
  const float* We2   = (const float*)d_in[7];  const float* be2   = (const float*)d_in[8];
  const float* Wc1   = (const float*)d_in[9];  const float* bc1   = (const float*)d_in[10];
  const float* Wc2   = (const float*)d_in[11]; const float* bc2   = (const float*)d_in[12];
  const float* bng   = (const float*)d_in[13]; const float* bnb   = (const float*)d_in[14];
  const float* Wih   = (const float*)d_in[15]; const float* Whh   = (const float*)d_in[16];
  const float* blstm = (const float*)d_in[17];
  const float* Wg1   = (const float*)d_in[18]; const float* bg1   = (const float*)d_in[19];
  const float* Wg2   = (const float*)d_in[20]; const float* bg2   = (const float*)d_in[21];
  const float* Wn1   = (const float*)d_in[22]; const float* bn1   = (const float*)d_in[23];
  const float* Wn2   = (const float*)d_in[24]; const float* bn2   = (const float*)d_in[25];
  const float* Wd1   = (const float*)d_in[26]; const float* bd1   = (const float*)d_in[27];
  const float* Wd2   = (const float*)d_in[28]; const float* bd2   = (const float*)d_in[29];
  const float* Wd3   = (const float*)d_in[30]; const float* bd3   = (const float*)d_in[31];
  const float* Wp1   = (const float*)d_in[32]; const float* bp1   = (const float*)d_in[33];
  const float* Wp2   = (const float*)d_in[34]; const float* bp2   = (const float*)d_in[35];
  const int*   ei    = (const int*)d_in[36];
  const int*   batch = (const int*)d_in[37];

  float* out        = (float*)d_out;
  float* latent     = out;                                   // B*64
  float* node_recon = out + (long)NGRAPH*64;                 // N*119
  float* elog       = node_recon + (long)NNODES*119;         // E
  float* erec       = elog + (long)NEDGES;                   // E*4

  float* W = (float*)d_ws;
  size_t o = 0;
  float* h      = W + o; o += (size_t)NNODES*64;
  float* z2     = W + o; o += (size_t)NNODES*64;
  float* agg    = W + o; o += (size_t)NNODES*64;
  float* colsum = W + o; o += 64;
  float* colsq  = W + o; o += 64;
  float* bnsc   = W + o; o += 64;
  float* bnsh   = W + o; o += 64;
  float* hs     = W + o; o += (size_t)NGRAPH*64;
  float* cs     = W + o; o += (size_t)NGRAPH*64;
  float* qstar  = W + o; o += (size_t)NGRAPH*128;
  float* gates  = W + o; o += (size_t)NGRAPH*256;
  float* att    = W + o; o += NNODES;
  float* wexp   = W + o; o += NNODES;
  unsigned* mkey = (unsigned*)(W + o); o += NGRAPH;
  float* den    = W + o; o += NGRAPH;
  float* rbuf   = W + o; o += (size_t)NGRAPH*64;
  float* ebuf   = W + o; o += (size_t)NEDGES*64;

  dim3 blk(128);
  int nodeWG = (NNODES + 63) / 64;
  int edgeWG = (NEDGES + 63) / 64;

  k_node_enc<<<nodeWG, blk, 0, stream>>>(x, feats, Wnode, bnode, h);
  k_edge_enc<<<edgeWG, blk, 0, stream>>>(ea, We1, be1, We2, be2, ebuf);

  for (int l = 0; l < 3; ++l){
    long n64 = (long)NNODES*64;
    k_zero<<<(int)((n64 + 255)/256), 256, 0, stream>>>(agg, n64);
    k_zero<<<1, 256, 0, stream>>>(colsum, 128);              // colsum+colsq contiguous
    k_message<<<(int)(((long)NEDGES*16 + 255)/256), 256, 0, stream>>>(h, ebuf, ei, agg);
    k_gine_mlp<<<nodeWG, blk, 0, stream>>>(h, agg, Wc1 + l*4096, bc1 + l*64,
                                           Wc2 + l*4096, bc2 + l*64, z2, colsum, colsq);
    k_bn_fin<<<1, 64, 0, stream>>>(colsum, colsq, bng + l*64, bnb + l*64, bnsc, bnsh);
    k_bn_apply<<<(int)((n64 + 255)/256), 256, 0, stream>>>(z2, bnsc, bnsh, h);
  }

  // Set2Set: zero hs, cs, qstar (contiguous), then 3 processing steps
  k_zero<<<(int)(((long)NGRAPH*256 + 255)/256), 256, 0, stream>>>(hs, (long)NGRAPH*256);
  for (int it = 0; it < 3; ++it){
    k_lstm_gates<<<NGRAPH, 256, 0, stream>>>(qstar, hs, Wih, Whh, blstm, gates);
    k_lstm_update<<<(NGRAPH*64 + 255)/256, 256, 0, stream>>>(gates, cs, hs);
    k_zero<<<((NGRAPH*66) + 255)/256, 256, 0, stream>>>((float*)mkey, NGRAPH*66); // mkey+den+rbuf
    k_att<<<(NNODES + 255)/256, 256, 0, stream>>>(h, hs, batch, att, mkey);
    k_wden<<<(NNODES + 255)/256, 256, 0, stream>>>(att, batch, mkey, wexp, den);
    k_rsum<<<(int)(((long)NNODES*16 + 255)/256), 256, 0, stream>>>(h, wexp, den, batch, rbuf);
    k_qstar<<<(NGRAPH*128 + 255)/256, 256, 0, stream>>>(hs, rbuf, qstar);
  }

  k_graph_mlp<<<(NGRAPH + 63)/64, blk, 0, stream>>>(qstar, Wg1, bg1, Wg2, bg2, latent);
  k_node_recon<<<nodeWG, blk, 0, stream>>>(latent, batch, Wn1, bn1, Wn2, bn2, node_recon);
  k_edge_logit<<<edgeWG, blk, 0, stream>>>(latent, batch, ei, Wp1, bp1, Wp2, bp2, elog);
  k_edge_recon<<<edgeWG, blk, 0, stream>>>(latent, batch, ei, Wd1, bd1, Wd2, bd2, Wd3, bd3, erec);
}
